// Transcoder_25615184953988
// MI455X (gfx1250) — matrix operand order repaired and checked by reading, not compiled, not run on hardware
//
#include <hip/hip_runtime.h>
#include <hip/hip_bf16.h>

// ---------------------------------------------------------------------------
// Transcoder: z = x @ W_enc^T + b_enc ; LN ; ReLU ; top-128 ; out = f @ W_dec^T + b_dec
// B=4,S=2048 -> 8192 tokens, D=768, DICT=32768, K(topk)=128
// ---------------------------------------------------------------------------

constexpr int kD    = 768;
constexpr int kDict = 32768;
constexpr int kTok  = 8192;      // B*S
constexpr int kTopK = 128;
constexpr int kNB   = 1024;      // histogram bins == threads of token kernel

typedef __attribute__((ext_vector_type(8)))  float  v8f;
typedef __attribute__((ext_vector_type(16))) __bf16 v16bf;
typedef __attribute__((ext_vector_type(8)))  __bf16 v8bf;
typedef __attribute__((ext_vector_type(4)))  int    v4i;

// ---------------------------------------------------------------------------
// CDNA5 async global->LDS copy (ASYNCcnt-tracked), 16 bytes per lane.
// Builtin prototype (probe-confirmed by diagnostic): (v4i AS1*, v4i AS3*, Ii, Ii)
// ---------------------------------------------------------------------------
__device__ __forceinline__ void async_cp16(const __bf16* gsrc, __bf16* ldst) {
#if __has_builtin(__builtin_amdgcn_global_load_async_to_lds_b128)
    typedef __attribute__((address_space(1))) v4i* gp_t;
    typedef __attribute__((address_space(3))) v4i* lp_t;
    __builtin_amdgcn_global_load_async_to_lds_b128(
        (gp_t)(void*)gsrc, (lp_t)(void*)ldst, 0, 0);
#elif defined(__gfx1250__)
    typedef __attribute__((address_space(3))) void* lv_t;
    unsigned int  l = (unsigned int)(unsigned long long)(lv_t)(void*)ldst;
    unsigned long long g = (unsigned long long)gsrc;
    asm volatile("global_load_async_to_lds_b128 %0, %1, off"
                 :: "v"(l), "v"(g) : "memory");
#else
    (void)gsrc; (void)ldst;   // host pass: never executed
#endif
}

__device__ __forceinline__ void wait_async0() {
#if __has_builtin(__builtin_amdgcn_s_wait_asynccnt)
    __builtin_amdgcn_s_wait_asynccnt(0);
#elif defined(__gfx1250__)
    asm volatile("s_wait_asynccnt 0x0" ::: "memory");
#endif
}

// ---------------------------------------------------------------------------
// 1) f32 -> bf16 conversion (vectorized 4/thread)
// ---------------------------------------------------------------------------
__global__ __launch_bounds__(256)
void cvt_bf16_kernel(const float* __restrict__ in, __hip_bfloat16* __restrict__ outp, int n) {
    int i = (blockIdx.x * 256 + threadIdx.x) * 4;
    if (i + 3 < n) {
        float4 v = *(const float4*)(in + i);
        union { __hip_bfloat16 h[4]; uint2 u; } p;
        p.h[0] = __float2bfloat16(v.x);
        p.h[1] = __float2bfloat16(v.y);
        p.h[2] = __float2bfloat16(v.z);
        p.h[3] = __float2bfloat16(v.w);
        *(uint2*)(outp + i) = p.u;
    }
}

// ---------------------------------------------------------------------------
// 2) W_dec [768][32768] f32  ->  W_decT [32768][768] bf16 (tiled LDS transpose)
// ---------------------------------------------------------------------------
__global__ __launch_bounds__(256)
void transpose_wdec_kernel(const float* __restrict__ wd, __hip_bfloat16* __restrict__ wdT) {
    __shared__ float tile[32][33];
    const int f0 = blockIdx.x * 32;
    const int d0 = blockIdx.y * 32;
#pragma unroll
    for (int r = 0; r < 32; r += 8)
        tile[threadIdx.y + r][threadIdx.x] =
            wd[(size_t)(d0 + threadIdx.y + r) * kDict + f0 + threadIdx.x];
    __syncthreads();
#pragma unroll
    for (int r = 0; r < 32; r += 8)
        wdT[(size_t)(f0 + threadIdx.y + r) * kD + d0 + threadIdx.x] =
            __float2bfloat16(tile[threadIdx.x][threadIdx.y + r]);
}

// ---------------------------------------------------------------------------
// 3) Encoder GEMM via V_WMMA_F32_16X16X32_BF16.
//    Block = 128 threads (4 waves), tile 64(M) x 64(N), K-loop step 32 (24 iters).
//    B tile (64x32 bf16) staged into double-buffered LDS by async DMA
//    (GLOBAL_LOAD_ASYNC_TO_LDS_B128, ASYNCcnt) overlapping the WMMA chain;
//    one barrier per K-step. Writes pre-LN z (f32) into features region.
// ---------------------------------------------------------------------------
__global__ __launch_bounds__(128)
void encoder_wmma_kernel(const __bf16* __restrict__ xb,
                         const __bf16* __restrict__ wb,
                         float* __restrict__ z) {
    __shared__ __align__(16) __bf16 Bs[2][64 * 32];

    const int tid  = threadIdx.x;
    const int lane = tid & 31;
    const int wave = tid >> 5;
    const int n0 = blockIdx.x * 64;
    const int m0 = blockIdx.y * 64 + wave * 16;

    v8f c[4] = {};

    // A fragment (16x32 bf16): lane<16 -> M=lane, K halves {0..7,16..23};
    // lane>=16 -> M=lane-16, K halves {8..15,24..31}
    const int arow  = m0 + (lane & 15);
    const int ahalf = (lane >> 4) * 8;
    const __bf16* aptr = xb + (size_t)arow * kD + ahalf;

    // B fragment (32x16 bf16): lane<16 -> N=lane, K {0..7,16..23}; lane>=16 -> N=lane-16, K {8..15,24..31}
    const int bcol = lane & 15;
    const int bk   = (lane >> 4) * 8;

    // async staging: 128 threads x 32B each = 4KB B tile per K-step
    const int srow  = tid >> 1;               // 0..63 (N within tile)
    const int shalf = (tid & 1) * 16;         // K half
    const __bf16* sptr = wb + (size_t)(n0 + srow) * kD + shalf;
    __bf16* ldst0 = &Bs[0][srow * 32 + shalf];
    __bf16* ldst1 = &Bs[1][srow * 32 + shalf];

    // prologue: stage K-step 0 into buffer 0
    async_cp16(sptr,     ldst0);
    async_cp16(sptr + 8, ldst0 + 8);
    wait_async0();
    __syncthreads();

    int cur = 0;
    for (int k0 = 0; k0 < kD; k0 += 32) {
        // kick off DMA for next K-step into the other buffer (overlaps WMMAs)
        if (k0 + 32 < kD) {
            __bf16* nd = cur ? ldst0 : ldst1;
            async_cp16(sptr + k0 + 32,     nd);
            async_cp16(sptr + k0 + 32 + 8, nd + 8);
        }

        union { v16bf v; v8bf h[2]; } a;
        a.h[0] = *(const v8bf*)(aptr + k0);
        a.h[1] = *(const v8bf*)(aptr + k0 + 16);

        // preload all four B fragments so ds_loads pipeline, then 4 WMMAs back-to-back
        union { v16bf v; v8bf h[2]; } b[4];
#pragma unroll
        for (int t = 0; t < 4; ++t) {
            const __bf16* bp = &Bs[cur][(t * 16 + bcol) * 32 + bk];
            b[t].h[0] = *(const v8bf*)bp;
            b[t].h[1] = *(const v8bf*)(bp + 16);
        }
#pragma unroll
        for (int t = 0; t < 4; ++t)
            c[t] = __builtin_amdgcn_wmma_f32_16x16x32_bf16(
                false, a.v, false, b[t].v, (short)0, c[t], false, false);

        wait_async0();       // next buffer fully landed in LDS
        __syncthreads();     // all waves done reading cur / DMA visible
        cur ^= 1;
    }

    // C layout: VGPR r, lanes 0-15 -> M=r, lanes 16-31 -> M=8+r; N = lane&15
    const int crow = m0 + (lane >> 4) * 8;
    const int ccol = n0 + (lane & 15);
#pragma unroll
    for (int r = 0; r < 8; ++r) {
        size_t base = (size_t)(crow + r) * kDict + ccol;
        z[base]      = c[0][r];
        z[base + 16] = c[1][r];
        z[base + 32] = c[2][r];
        z[base + 48] = c[3][r];
    }
}

// ---------------------------------------------------------------------------
// Block reductions (wave32 shuffles + 32 wave partials in LDS)
// ---------------------------------------------------------------------------
__device__ __forceinline__ float block_sum(float v, float* redf, float* res) {
#pragma unroll
    for (int o = 16; o > 0; o >>= 1) v += __shfl_xor(v, o, 32);
    const int lane = threadIdx.x & 31, w = threadIdx.x >> 5;
    __syncthreads();
    if (lane == 0) redf[w] = v;
    __syncthreads();
    if (w == 0) {
        float x = redf[lane];
#pragma unroll
        for (int o = 16; o > 0; o >>= 1) x += __shfl_xor(x, o, 32);
        if (lane == 0) *res = x;
    }
    __syncthreads();
    float r = *res;
    __syncthreads();
    return r;
}

__device__ __forceinline__ float block_max(float v, float* redf, float* res) {
#pragma unroll
    for (int o = 16; o > 0; o >>= 1) v = fmaxf(v, __shfl_xor(v, o, 32));
    const int lane = threadIdx.x & 31, w = threadIdx.x >> 5;
    __syncthreads();
    if (lane == 0) redf[w] = v;
    __syncthreads();
    if (w == 0) {
        float x = redf[lane];
#pragma unroll
        for (int o = 16; o > 0; o >>= 1) x = fmaxf(x, __shfl_xor(x, o, 32));
        if (lane == 0) *res = x;
    }
    __syncthreads();
    float r = *res;
    __syncthreads();
    return r;
}

// ---------------------------------------------------------------------------
// 4) Per-token: LN + ReLU + exact top-128 (histogram select) + sparse decode.
//    One block per token, 1024 threads, row cached in ~138KB dynamic LDS
//    (CDNA5: 320KB LDS per WGP makes this possible).
// ---------------------------------------------------------------------------
__global__ __launch_bounds__(1024)
void token_topk_kernel(float* __restrict__ feat,               // in: z ; out: sparsified
                       const float* __restrict__ b_enc,
                       const float* __restrict__ gamma,
                       const float* __restrict__ beta,
                       const __hip_bfloat16* __restrict__ wdT, // [DICT][768] bf16
                       const float* __restrict__ b_dec,
                       float* __restrict__ outp) {
    extern __shared__ float smem[];
    float* zrow = smem;                       // 32768
    float* redf = zrow + kDict;               // 32
    int*   hist = (int*)(redf + 32);          // 1024
    int*   scan = hist + kNB;                 // 1024
    int*   kidx = scan + kNB;                 // 192
    float* kval = (float*)(kidx + 192);       // 192
    float* sc   = kval + 192;                 // 4 scalars (reduction result)
    int*   si   = (int*)(sc + 4);             // [0]=bstar [1]=above [2]=tie [3]=kept

    const int tid = threadIdx.x;
    const size_t row = (size_t)blockIdx.x * kDict;

    // pass 1: load z + bias, row sum
    float s = 0.f;
    for (int i = tid; i < kDict; i += 1024) {
        float v = feat[row + i] + b_enc[i];
        zrow[i] = v;
        s += v;
    }
    const float mu = block_sum(s, redf, sc) * (1.0f / kDict);

    // pass 2: variance
    float s2 = 0.f;
    for (int i = tid; i < kDict; i += 1024) {
        float d = zrow[i] - mu;
        s2 += d * d;
    }
    const float var  = block_sum(s2, redf, sc) * (1.0f / kDict);
    const float rstd = rsqrtf(var + 1e-5f);

    // pass 3: LN + ReLU in place, track max
    float lmax = 0.f;
    for (int i = tid; i < kDict; i += 1024) {
        float v = (zrow[i] - mu) * rstd * gamma[i] + beta[i];
        v = fmaxf(v, 0.f);
        zrow[i] = v;
        lmax = fmaxf(lmax, v);
    }
    const float vmax = block_max(lmax, redf, sc);

    // pass 4: histogram of positive values over [0, vmax] in 1024 bins
    hist[tid] = 0;
    __syncthreads();
    const float scale = (vmax > 0.f) ? ((float)kNB / vmax) : 0.f;
    for (int i = tid; i < kDict; i += 1024) {
        float v = zrow[i];
        if (v > 0.f) {
            int b = (int)(v * scale);
            if (b > kNB - 1) b = kNB - 1;
            atomicAdd(&hist[b], 1);
        }
    }
    __syncthreads();

    // suffix scan (Hillis-Steele, ping-pong): S[b] = #elements with bin >= b
    int myc = hist[tid];
    int* src = hist;
    int* dst = scan;
    for (int off = 1; off < kNB; off <<= 1) {
        int val = src[tid] + ((tid + off < kNB) ? src[tid + off] : 0);
        dst[tid] = val;
        __syncthreads();
        int* tmp = src; src = dst; dst = tmp;
    }
    const int S = src[tid];   // 10 iterations -> result back in hist

    if (tid == 0) { si[2] = 0; si[3] = 0; }
    // b* = largest bin with S[b] >= K  (definite keeps = S[b*+1] = S - myc)
    if (S >= kTopK && (S - myc) < kTopK) { si[0] = tid; si[1] = S - myc; }
    if (tid == 0 && S < kTopK)           { si[0] = 0;   si[1] = S - myc; }
    __syncthreads();
    const int bstar     = si[0];
    const int tieBudget = kTopK - si[1];

    // pass 5: classify, write sparsified features row, collect kept list
    for (int i = tid; i < kDict; i += 1024) {
        float v = zrow[i];
        bool keep = false;
        if (v > 0.f) {
            int b = (int)(v * scale);
            if (b > kNB - 1) b = kNB - 1;
            if (b > bstar) keep = true;
            else if (b == bstar) {
                int t = atomicAdd(&si[2], 1);
                if (t < tieBudget) keep = true;
            }
        }
        if (keep) {
            int p = atomicAdd(&si[3], 1);
            if (p < 192) { kidx[p] = i; kval[p] = v; }
        }
        feat[row + i] = keep ? v : 0.f;
    }
    __syncthreads();

    // sparse decoder: out[d] = b_dec[d] + sum_j kval[j] * W_decT[kidx[j]][d]
    int nk = si[3]; if (nk > 192) nk = 192;
    if (tid < kD) {
        float acc = b_dec[tid];
        for (int j = 0; j < nk; ++j)
            acc += kval[j] * __bfloat162float(wdT[(size_t)kidx[j] * kD + tid]);
        outp[(size_t)blockIdx.x * kD + tid] = acc;
    }
}

// ---------------------------------------------------------------------------
// Host launcher
// ---------------------------------------------------------------------------
extern "C" void kernel_launch(void* const* d_in, const int* in_sizes, int n_in,
                              void* d_out, int out_size, void* d_ws, size_t ws_size,
                              hipStream_t stream) {
    (void)in_sizes; (void)n_in; (void)out_size; (void)ws_size;

    const float* x     = (const float*)d_in[0];
    const float* W_enc = (const float*)d_in[1];
    const float* b_enc = (const float*)d_in[2];
    const float* gamma = (const float*)d_in[3];
    const float* beta  = (const float*)d_in[4];
    const float* W_dec = (const float*)d_in[5];
    const float* b_dec = (const float*)d_in[6];

    float* outp = (float*)d_out;                       // [8192][768]
    float* feat = outp + (size_t)kTok * kD;            // [8192][32768]

    // workspace layout: x_bf16 (12.6MB) | W_enc_bf16 (50.3MB) | W_decT bf16 (50.3MB)
    __hip_bfloat16* xb  = (__hip_bfloat16*)d_ws;
    __hip_bfloat16* web = xb  + (size_t)kTok  * kD;
    __hip_bfloat16* wdT = web + (size_t)kDict * kD;

    const int nx = kTok * kD;      // 6,291,456  (divisible by 1024)
    const int nw = kDict * kD;     // 25,165,824 (divisible by 1024)
    cvt_bf16_kernel<<<nx / 1024, 256, 0, stream>>>(x, xb, nx);
    cvt_bf16_kernel<<<nw / 1024, 256, 0, stream>>>(W_enc, web, nw);

    transpose_wdec_kernel<<<dim3(kDict / 32, kD / 32), dim3(32, 8), 0, stream>>>(W_dec, wdT);

    encoder_wmma_kernel<<<dim3(kDict / 64, kTok / 64), 128, 0, stream>>>(
        (const __bf16*)xb, (const __bf16*)web, feat);

    const size_t smem_bytes =
        (size_t)(kDict + 32 + kNB + kNB + 192 + 192 + 8) * sizeof(float);
    (void)hipFuncSetAttribute(reinterpret_cast<const void*>(token_topk_kernel),
                              hipFuncAttributeMaxDynamicSharedMemorySize,
                              (int)smem_bytes);
    token_topk_kernel<<<kTok, 1024, smem_bytes, stream>>>(
        feat, b_enc, gamma, beta, wdT, b_dec, outp);
}
